// EQGATEdgeGNN_55783035240747
// MI455X (gfx1250) — matrix operand order repaired and checked by reading, not compiled, not run on hardware
//
#include <hip/hip_runtime.h>

typedef __attribute__((ext_vector_type(16))) __bf16 v16bf;
typedef __attribute__((ext_vector_type(8)))  float  v8f;

static constexpr int kN = 25000, kE = 400000, kG = 512;
static constexpr int kNP = ((kN + 31) / 32) * 32;   // node count padded to full 2-wave blocks
static constexpr float kEPS = 1e-6f;
static_assert(kE % 32 == 0, "edge kernel assumes full 16-edge tiles, 2 per block");
static_assert(kNP % 32 == 0 && kNP >= kN, "padded node tiles");

#define DEV __device__ __forceinline__

DEV unsigned short f2bf(float x){
  unsigned u = __float_as_uint(x);
  unsigned r = u + 0x7FFFu + ((u >> 16) & 1u);
  return (unsigned short)(r >> 16);
}
DEV float dev_silu(float x){ return x * (1.0f / (1.0f + __expf(-x))); }

DEV v8f wmma_bf(v16bf a, v16bf b, v8f c){
  return __builtin_amdgcn_wmma_f32_16x16x32_bf16(false, a, false, b, (short)0, c, false, false);
}

// A fragment (16x32 bf16) from row-major LDS tile. Per CDNA5 ISA layout:
// lane m=l&15 is row M; lane group g=l>>4 selects K halves {0-7,16-23} vs {8-15,24-31}.
DEV v16bf load_A_lds(const unsigned short* base, int ldk, int k0, int lane){
  int m = lane & 15, g = lane >> 4;
  v16bf a; unsigned short* ap = (unsigned short*)&a;
  const unsigned short* row = base + m * ldk + k0;
#pragma unroll
  for (int p = 0; p < 8; p++){
    int k = (p < 4) ? (g * 8 + 2 * p) : (16 + g * 8 + 2 * (p - 4));
    ap[2 * p]     = row[k];
    ap[2 * p + 1] = row[k + 1];
  }
  return a;
}
// B fragment: weights pre-swizzled so each lane's 16 elements are contiguous (32B load).
DEV v16bf load_B_frag(const unsigned short* wf, int ktiles, int kt, int nt, int lane){
  return *(const v16bf*)(wf + ((nt * ktiles + kt) * 32 + lane) * 16);
}

// ---------------- small utility kernels ----------------
__global__ void k_zero(float* p, int n){
  int i = blockIdx.x * blockDim.x + threadIdx.x, st = gridDim.x * blockDim.x;
  for (; i < n; i += st) p[i] = 0.f;
}
__global__ void k_deg(const int* tgt, float* deg){
  int e = blockIdx.x * blockDim.x + threadIdx.x;
  if (e < kE) atomicAdd(&deg[tgt[e]], 1.0f);
}
__global__ void k_edge_geom(const float* p, const int* src, const int* tgt,
                            float* dbuf, float* abuf, float* rbuf){
  int e = blockIdx.x * blockDim.x + threadIdx.x;
  if (e >= kE) return;
  int si = src[e], ti = tgt[e];
  float ps0=p[si*3],ps1=p[si*3+1],ps2=p[si*3+2];
  float pt0=p[ti*3],pt1=p[ti*3+1],pt2=p[ti*3+2];
  float r0=pt0-ps0, r1=pt1-ps1, r2=pt2-ps2;
  float nsv = sqrtf(ps0*ps0+ps1*ps1+ps2*ps2);
  float ntv = sqrtf(pt0*pt0+pt1*pt1+pt2*pt2);
  float is_ = (nsv != 0.f) ? 1.f/nsv : 0.f;
  float it_ = (ntv != 0.f) ? 1.f/ntv : 0.f;
  float a = (pt0*it_)*(ps0*is_) + (pt1*it_)*(ps1*is_) + (pt2*it_)*(ps2*is_);
  float d = sqrtf(fmaxf(r0*r0+r1*r1+r2*r2, kEPS));
  float inv1d = 1.f/(1.f+d);
  dbuf[e]=d; abuf[e]=a;
  rbuf[e*3]=r0*inv1d; rbuf[e*3+1]=r1*inv1d; rbuf[e*3+2]=r2*inv1d;
}

// per-graph stats: [0..63] sum(s), [64..127] sum(s^2), [128] count, [129] sum(|v|^2/VI)
__global__ void k_gstat(const float* s, const float* v, const int* batch, float* gstat){
  int n = blockIdx.x * blockDim.x + threadIdx.x;
  if (n >= kN) return;
  float* gs = gstat + batch[n] * 130;
  for (int c = 0; c < 64; c++){
    float x = s[n*64+c];
    atomicAdd(&gs[c], x); atomicAdd(&gs[64+c], x*x);
  }
  float v2 = 0.f;
  for (int j = 0; j < 48; j++){ float x = v[n*48+j]; v2 += x*x; }
  atomicAdd(&gs[128], 1.0f); atomicAdd(&gs[129], v2 * (1.0f/16.0f));
}
__global__ void k_gnorm(const float* s, const float* v, const int* batch, const float* gstat,
                        const float* w, const float* b,
                        float* s_n, unsigned short* s_nbf, float* v_n){
  int n = blockIdx.x * blockDim.x + threadIdx.x;
  if (n >= kN) return;
  const float* gs = gstat + batch[n] * 130;
  float inv = 1.0f / fmaxf(gs[128], 1.0f);
  float vscale = 1.0f / sqrtf(gs[129]*inv + kEPS);
  for (int c = 0; c < 64; c++){
    float mean = gs[c]*inv;
    float var = fmaxf(gs[64+c]*inv - mean*mean, 0.0f);
    float y = (s[n*64+c]-mean) * (1.0f/sqrtf(var+kEPS)) * w[c] + b[c];
    s_n[n*64+c] = y; s_nbf[n*64+c] = f2bf(y);
  }
  for (int j = 0; j < 48; j++) v_n[n*48+j] = v[n*48+j]*vscale;
}

// ---------------- weight packing into B-fragment order ----------------
struct PackEnt { const float* src; int K, Nn, Kt, Nt, off; };
struct PackArgs { PackEnt e[10]; int count; unsigned short* dst; };
__global__ void k_pack(PackArgs pa){
  int tid = blockIdx.x * blockDim.x + threadIdx.x, nth = gridDim.x * blockDim.x;
  for (int q = 0; q < pa.count; q++){
    const float* W = pa.e[q].src;
    int K = pa.e[q].K, Nn = pa.e[q].Nn, ktiles = pa.e[q].Kt;
    unsigned short* dst = pa.dst + pa.e[q].off;
    int total = ktiles * pa.e[q].Nt * 512;
    for (int idx = tid; idx < total; idx += nth){
      int el = idx & 15, lane = (idx >> 4) & 31, tile = idx >> 9;
      int nt = tile / ktiles, kt = tile - nt * ktiles;
      int n = lane & 15, g = lane >> 4, p = el >> 1, wbit = el & 1;
      int k = kt*32 + ((p < 4) ? (g*8 + 2*p) : (16 + g*8 + 2*(p - 4))) + wbit;
      int ncol = nt*16 + n;
      float val = (k < K && ncol < Nn) ? W[k*Nn + ncol] : 0.0f;
      dst[idx] = f2bf(val);
    }
  }
}

// ---------------- node pre-kernel: ns = s_n @ Ws + bs ; vv = v_n @ Wv ----------------
// All per-node buffers are padded to kNP rows -> every tile is full, no guards.
struct NodePreArgs {
  const unsigned short* s_nbf; const float* v_n;
  const unsigned short* fWs; const float* bs;
  const unsigned short* fWv; int has_v;
  float* ns; float* vv;
};
__global__ __launch_bounds__(64) void k_nodepre(NodePreArgs A){
  __shared__ __align__(16) struct { unsigned short ssm[16][64]; unsigned short vsm[3][16][32]; } sm[2];
  int wv = threadIdx.x >> 5, lane = threadIdx.x & 31;
  auto& S = sm[wv];
  int n0 = (blockIdx.x * 2 + wv) * 16;
  {
    int row = lane & 15, hh = lane >> 4;
    const uint4* q = (const uint4*)(A.s_nbf + (n0 + row)*64 + hh*32);
    uint4* d = (uint4*)&S.ssm[row][hh*32];
#pragma unroll
    for (int j = 0; j < 4; j++) d[j] = q[j];
    if (A.has_v){
      for (int idx = lane; idx < 768; idx += 32){
        int dim = idx >> 8, rem = idx & 255, r = rem >> 4, k = rem & 15;
        S.vsm[dim][r][k] = f2bf(A.v_n[(n0 + r)*48 + dim*16 + k]);
        S.vsm[dim][r][16+k] = 0;
      }
    }
  }
  __syncthreads();
  {
    int n = lane & 15, g = lane >> 4;
#pragma unroll
    for (int nt = 0; nt < 4; nt++){
      v8f acc = {};
#pragma unroll
      for (int kt = 0; kt < 2; kt++)
        acc = wmma_bf(load_A_lds(&S.ssm[0][0], 64, kt*32, lane),
                      load_B_frag(A.fWs, 2, kt, nt, lane), acc);
      float bb = A.bs[nt*16+n];
      float* dst = A.ns + (size_t)(n0 + 8*g)*64 + nt*16 + n;
#pragma unroll
      for (int r = 0; r < 8; r++) dst[r*64] = acc[r] + bb;
    }
    if (A.has_v){
#pragma unroll
      for (int dim = 0; dim < 3; dim++){
        v8f acc = {};
        acc = wmma_bf(load_A_lds(&S.vsm[dim][0][0], 32, 0, lane),
                      load_B_frag(A.fWv, 1, 0, 0, lane), acc);
        float* dst = A.vv + (size_t)(n0 + 8*g)*48 + dim*16 + n;
#pragma unroll
        for (int r = 0; r < 8; r++) dst[r*48] = acc[r];
      }
    }
  }
}

// ---------------- fused edge kernel (all tiles full: kE % 32 == 0) ----------------
struct EdgeArgs {
  const int *src, *tgt;
  const float *dbuf, *abuf, *rbuf, *ohe, *gemb;
  const unsigned short *s_nbf;
  float *e_a;
  const float *ns, *vv;
  float *s_sum, *v_sum, *p_sum;
  const unsigned short *fWpre, *fW1, *fW2, *fWpost;
  const float *be_pre, *be1, *be2, *be_post;
  int has_v, out2;
};
__global__ __launch_bounds__(64) void k_edge(EdgeArgs A){
  __shared__ __align__(16) struct {
    unsigned short feat[16][192];
    unsigned short h[16][64];
    float o[16][128];
    unsigned short ebf[16][32];
    int srcI[16], tgtI[16];
    float dD[16], aA[16], cC[16];
  } sm[2];
  int wv = threadIdx.x >> 5, lane = threadIdx.x & 31;
  auto& S = sm[wv];
  int e0 = (blockIdx.x * 2 + wv) * 16;

  if (lane < 16){
    int e = e0 + lane;
    S.srcI[lane] = A.src[e]; S.tgtI[lane] = A.tgt[e];
    float d = A.dbuf[e]; S.dD[lane] = d; S.aA[lane] = A.abuf[e];
    float c = 0.5f * (__cosf(0.6283185307f * d) + 1.0f);
    S.cC[lane] = (d <= 5.0f) ? c : 0.0f;
  }
  {
    for (int idx = lane; idx < 256; idx += 32){
      int row = idx >> 4, col = idx & 15;
      S.ebf[row][col] = f2bf(A.e_a[(e0 + row)*16 + col]);
      S.ebf[row][16+col] = 0;
    }
  }
  __syncthreads();
  {
    // de0 = e @ We_pre + be_pre  -> feat[:,128:144]
    v8f acc = {};
    acc = wmma_bf(load_A_lds(&S.ebf[0][0], 32, 0, lane),
                  load_B_frag(A.fWpre, 1, 0, 0, lane), acc);
    int n = lane & 15, g = lane >> 4;
#pragma unroll
    for (int r = 0; r < 8; r++) S.feat[r + 8*g][128 + n] = f2bf(acc[r] + A.be_pre[n]);
    // gather s[tgt], s[src] (bf16, 128B rows, uint4 vectorized)
    int row = lane & 15, hh = lane >> 4;
    const uint4* qt = (const uint4*)(A.s_nbf + S.tgtI[row]*64 + hh*32);
    const uint4* qs = (const uint4*)(A.s_nbf + S.srcI[row]*64 + hh*32);
    uint4* ft = (uint4*)&S.feat[row][hh*32];
    uint4* fs = (uint4*)&S.feat[row][64 + hh*32];
#pragma unroll
    for (int j = 0; j < 4; j++){ ft[j] = qt[j]; fs[j] = qs[j]; }
    if (lane < 16){
      int e = e0 + row;
      S.feat[row][144] = f2bf(S.dD[row]);
      S.feat[row][145] = f2bf(S.aA[row]);
      for (int j = 0; j < 5; j++)  S.feat[row][146+j] = f2bf(A.ohe[e*5+j]);
      for (int j = 0; j < 16; j++) S.feat[row][151+j] = f2bf(A.gemb[e*16+j]);
      for (int j = 167; j < 192; j++) S.feat[row][j] = 0;
    }
  }
  __syncthreads();
  { // h = silu(feat @ We1 + be1)
#pragma unroll
    for (int nt = 0; nt < 4; nt++){
      v8f acc = {};
#pragma unroll
      for (int kt = 0; kt < 6; kt++)
        acc = wmma_bf(load_A_lds(&S.feat[0][0], 192, kt*32, lane),
                      load_B_frag(A.fW1, 6, kt, nt, lane), acc);
      int n = lane & 15, g = lane >> 4;
      float bb = A.be1[nt*16+n];
#pragma unroll
      for (int r = 0; r < 8; r++) S.h[r + 8*g][nt*16 + n] = f2bf(dev_silu(acc[r] + bb));
    }
  }
  __syncthreads();
  { // o = h @ We2 + be2
#pragma unroll
    for (int nt = 0; nt < 8; nt++){
      v8f acc = {};
#pragma unroll
      for (int kt = 0; kt < 2; kt++)
        acc = wmma_bf(load_A_lds(&S.h[0][0], 64, kt*32, lane),
                      load_B_frag(A.fW2, 2, kt, nt, lane), acc);
      int n = lane & 15, g = lane >> 4, col = nt*16 + n;
      float bb = (col < A.out2) ? A.be2[col] : 0.0f;
#pragma unroll
      for (int r = 0; r < 8; r++) S.o[r + 8*g][col] = acc[r] + bb;
    }
  }
  __syncthreads();
  int basep = 64 + (A.has_v ? 32 : 16);
  {
    int row = lane & 15, hh = lane >> 4;
    int e = e0 + row;
    int tI = S.tgtI[row], sI = S.srcI[row];
    float C = S.cC[row];
    for (int j = 0; j < 32; j++){
      int col = hh*32 + j;
      atomicAdd(&A.s_sum[tI*64 + col], S.o[row][col] * A.ns[sI*64 + col] * C);
    }
    for (int j = 0; j < 24; j++){
      int k48 = hh*24 + j, dim = k48 >> 4, k = k48 & 15;
      float rr = A.rbuf[e*3 + dim];
      float mv = A.has_v
        ? (S.o[row][64+k] * A.vv[sI*48 + dim*16 + k] + S.o[row][80+k] * rr)
        : (S.o[row][64+k] * rr);
      atomicAdd(&A.v_sum[tI*48 + dim*16 + k], mv * C);
    }
    if (hh == 0){
      float gp = S.o[row][basep];
      for (int dim = 0; dim < 3; dim++)
        atomicAdd(&A.p_sum[tI*3 + dim], gp * A.rbuf[e*3 + dim]);
      // stage silu(ge) for e-update GEMM
      for (int j = 0; j < 16; j++)
        S.ebf[row][j] = f2bf(dev_silu(S.o[row][basep + 1 + j]));
    }
  }
  __syncthreads();
  { // e += silu(ge) @ We_post + be_post
    v8f acc = {};
    acc = wmma_bf(load_A_lds(&S.ebf[0][0], 32, 0, lane),
                  load_B_frag(A.fWpost, 1, 0, 0, lane), acc);
    int n = lane & 15, g = lane >> 4;
    float bb = A.be_post[n];
    float* dst = A.e_a + (size_t)(e0 + 8*g)*16 + n;
#pragma unroll
    for (int r = 0; r < 8; r++) dst[r*16] += acc[r] + bb;
  }
}

__global__ void k_node_update(const float* s_n, const float* v_n,
                              const float* s_sum, const float* v_sum, const float* p_sum,
                              const float* deg, float* s_a, float* v_a, float* p_a){
  int n = blockIdx.x * blockDim.x + threadIdx.x;
  if (n >= kN) return;
  float invd = 1.0f / fmaxf(deg[n], 1.0f);
  for (int c = 0; c < 64; c++) s_a[n*64+c] = s_n[n*64+c] + s_sum[n*64+c];
  for (int j = 0; j < 48; j++) v_a[n*48+j] = v_n[n*48+j] + v_sum[n*48+j]*invd;
  for (int d = 0; d < 3; d++)  p_a[n*3+d] += p_sum[n*3+d]*invd;
}

// ---------------- node MLP (use_mlp layers); buffers padded -> no guards ----------------
struct MlpArgs {
  float *s_a, *v_a;
  const unsigned short *fWu1, *fWu2, *fWg1, *fWg2;
  const float *bg1, *bg2;
};
__global__ __launch_bounds__(64) void k_node_mlp(MlpArgs M){
  __shared__ __align__(16) struct {
    unsigned short vsm[3][16][32];
    unsigned short xsm[16][96];
    unsigned short hsm[16][64];
    float oo[16][96];
  } sm[2];
  int wv = threadIdx.x >> 5, lane = threadIdx.x & 31;
  auto& S = sm[wv];
  int n0 = (blockIdx.x * 2 + wv) * 16;
  {
    for (int idx = lane; idx < 768; idx += 32){
      int dim = idx >> 8, rem = idx & 255, r = rem >> 4, k = rem & 15;
      S.vsm[dim][r][k] = f2bf(M.v_a[(n0 + r)*48 + dim*16 + k]);
      S.vsm[dim][r][16+k] = 0;
    }
    int row = lane & 15, hh = lane >> 4;
    for (int j = 0; j < 32; j++){ int c = hh*32 + j; S.xsm[row][c] = f2bf(M.s_a[(n0 + row)*64+c]); }
    if (hh == 0) for (int j = 0; j < 16; j++) S.xsm[row][80+j] = 0;
  }
  __syncthreads();
  { // vn = sqrt(sum_d (v@Wu1)^2 + eps) -> xsm[:,64:80]
    v8f sq = {};
#pragma unroll
    for (int dim = 0; dim < 3; dim++){
      v8f acc = {};
      acc = wmma_bf(load_A_lds(&S.vsm[dim][0][0], 32, 0, lane),
                    load_B_frag(M.fWu1, 1, 0, 0, lane), acc);
#pragma unroll
      for (int r = 0; r < 8; r++) sq[r] += acc[r]*acc[r];
    }
    int n = lane & 15, g = lane >> 4;
#pragma unroll
    for (int r = 0; r < 8; r++) S.xsm[r + 8*g][64 + n] = f2bf(sqrtf(sq[r] + kEPS));
  }
  __syncthreads();
  { // hh = silu([s,vn] @ Wg1 + bg1)
#pragma unroll
    for (int nt = 0; nt < 4; nt++){
      v8f acc = {};
#pragma unroll
      for (int kt = 0; kt < 3; kt++)
        acc = wmma_bf(load_A_lds(&S.xsm[0][0], 96, kt*32, lane),
                      load_B_frag(M.fWg1, 3, kt, nt, lane), acc);
      int n = lane & 15, g = lane >> 4;
      float bb = M.bg1[nt*16+n];
#pragma unroll
      for (int r = 0; r < 8; r++) S.hsm[r + 8*g][nt*16 + n] = f2bf(dev_silu(acc[r] + bb));
    }
  }
  __syncthreads();
  { // oo = hh @ Wg2 + bg2 (80 cols)
#pragma unroll
    for (int nt = 0; nt < 5; nt++){
      v8f acc = {};
#pragma unroll
      for (int kt = 0; kt < 2; kt++)
        acc = wmma_bf(load_A_lds(&S.hsm[0][0], 64, kt*32, lane),
                      load_B_frag(M.fWg2, 2, kt, nt, lane), acc);
      int n = lane & 15, g = lane >> 4, col = nt*16 + n;
      float bb = M.bg2[col];
#pragma unroll
      for (int r = 0; r < 8; r++) S.oo[r + 8*g][col] = acc[r] + bb;
    }
  }
  __syncthreads();
  {
    int row = lane & 15, hh = lane >> 4;
    for (int j = 0; j < 32; j++){ int c = hh*32 + j; M.s_a[(n0 + row)*64+c] += S.oo[row][c]; }
    // v += gate * (v @ Wu2)   (vsm still holds pre-update v)
    int n = lane & 15, g = lane >> 4;
#pragma unroll
    for (int dim = 0; dim < 3; dim++){
      v8f acc = {};
      acc = wmma_bf(load_A_lds(&S.vsm[dim][0][0], 32, 0, lane),
                    load_B_frag(M.fWu2, 1, 0, 0, lane), acc);
      float* dst = M.v_a + (size_t)(n0 + 8*g)*48 + dim*16 + n;
#pragma unroll
      for (int r = 0; r < 8; r++) dst[r*48] += S.oo[r + 8*g][64 + n] * acc[r];
    }
  }
}

// ---------------- host ----------------
struct LayerP {
  const float *We1,*We2,*We_post,*We_pre,*Wg1,*Wg2,*Ws,*Wu1,*Wu2,*Wv;
  const float *be1,*be2,*be_post,*be_pre,*bg1,*bg2,*bs,*ln_b,*ln_w;
};

enum { O_WPRE=0, O_W1=512, O_W2=12800, O_WPOST=20992, O_WS=21504,
       O_WV=25600, O_WU1=26112, O_WU2=26624, O_WG1=27136, O_WG2=33280 };

extern "C" void kernel_launch(void* const* d_in, const int* in_sizes, int n_in,
                              void* d_out, int out_size, void* d_ws, size_t ws_size,
                              hipStream_t stream){
  (void)out_size; (void)ws_size;
  // locate batch / edge_index by unique flat sizes
  int bi = -1, ei = -1;
  for (int i = 0; i < n_in; i++){
    if (in_sizes[i] == kN) bi = i;
    if (in_sizes[i] == 2*kE) ei = i;
  }
  if (bi < 0) bi = n_in - 2;
  if (ei < 0) ei = n_in - 1;
  const float* in_s = (const float*)d_in[0];
  const float* in_v = (const float*)d_in[1];
  const float* in_p = (const float*)d_in[2];
  const float* in_e = (const float*)d_in[3];
  const float* ohe  = (const float*)d_in[4];
  const float* gemb = (const float*)d_in[5];
  const int* batch = (const int*)d_in[bi];
  const int* src = (const int*)d_in[ei];
  const int* tgt = src + kE;

  // params (pytree flatten: dict keys sorted)
  LayerP Ls[5];
  int pi = 6;
  for (int i = 0; i < 5; i++){
    bool has_v = i > 0, mlp = i < 4;
    LayerP& L = Ls[i];
    L.We1 = (const float*)d_in[pi++]; L.We2 = (const float*)d_in[pi++];
    L.We_post = (const float*)d_in[pi++]; L.We_pre = (const float*)d_in[pi++];
    if (mlp){ L.Wg1 = (const float*)d_in[pi++]; L.Wg2 = (const float*)d_in[pi++]; }
    L.Ws = (const float*)d_in[pi++];
    if (mlp){ L.Wu1 = (const float*)d_in[pi++]; L.Wu2 = (const float*)d_in[pi++]; }
    if (has_v) L.Wv = (const float*)d_in[pi++];
    L.be1 = (const float*)d_in[pi++]; L.be2 = (const float*)d_in[pi++];
    L.be_post = (const float*)d_in[pi++]; L.be_pre = (const float*)d_in[pi++];
    if (mlp){ L.bg1 = (const float*)d_in[pi++]; L.bg2 = (const float*)d_in[pi++]; }
    L.bs = (const float*)d_in[pi++];
    L.ln_b = (const float*)d_in[pi++]; L.ln_w = (const float*)d_in[pi++];
  }
  const float* out_ln_b = (const float*)d_in[pi++];
  const float* out_ln_w = (const float*)d_in[pi++];

  // workspace layout (per-node buffers padded to kNP rows)
  char* wp = (char*)d_ws;
  auto alloc = [&](size_t bytes)->char*{ char* r = wp; wp += (bytes + 255) & ~(size_t)255; return r; };
  float* s_a  = (float*)alloc((size_t)kNP*64*4);
  float* v_a  = (float*)alloc((size_t)kNP*48*4);
  float* p_a  = (float*)alloc((size_t)kN*3*4);
  float* e_a  = (float*)alloc((size_t)kE*16*4);
  float* s_n  = (float*)alloc((size_t)kNP*64*4);
  float* v_n  = (float*)alloc((size_t)kNP*48*4);
  unsigned short* s_nbf = (unsigned short*)alloc((size_t)kNP*64*2);
  float* ns   = (float*)alloc((size_t)kNP*64*4);
  float* vv   = (float*)alloc((size_t)kNP*48*4);
  float* dbuf = (float*)alloc((size_t)kE*4);
  float* abuf = (float*)alloc((size_t)kE*4);
  float* rbuf = (float*)alloc((size_t)kE*3*4);
  float* deg  = (float*)alloc((size_t)kN*4);
  float* gstat= (float*)alloc((size_t)kG*130*4);
  float* s_sum= (float*)alloc((size_t)kN*115*4);   // s_sum | v_sum | p_sum contiguous
  float* v_sum= s_sum + (size_t)kN*64;
  float* p_sum= v_sum + (size_t)kN*48;
  unsigned short* wfrag = (unsigned short*)alloc(38400*2);

  float* out = (float*)d_out;
  float* out_s = out;
  float* out_v = out + (size_t)kN*64;
  float* out_e = out_v + (size_t)kN*48;
  float* out_p = out_e + (size_t)kE*16;

  const int NB = (kN + 255)/256, EB = (kE + 255)/256;
  const int nodeBlk = kNP / 32;
  const int edgeBlk = kE / 32;

  hipMemcpyAsync(s_a, in_s, (size_t)kN*64*4, hipMemcpyDeviceToDevice, stream);
  hipMemcpyAsync(v_a, in_v, (size_t)kN*48*4, hipMemcpyDeviceToDevice, stream);
  hipMemcpyAsync(p_a, in_p, (size_t)kN*3*4,  hipMemcpyDeviceToDevice, stream);
  hipMemcpyAsync(e_a, in_e, (size_t)kE*16*4, hipMemcpyDeviceToDevice, stream);
  // init padded tail rows of node buffers (poisoned ws otherwise)
  k_zero<<<64, 256, 0, stream>>>(s_a + (size_t)kN*64, (kNP-kN)*64);
  k_zero<<<64, 256, 0, stream>>>(v_a + (size_t)kN*48, (kNP-kN)*48);
  k_zero<<<64, 256, 0, stream>>>(s_n + (size_t)kN*64, (kNP-kN)*64);
  k_zero<<<64, 256, 0, stream>>>(v_n + (size_t)kN*48, (kNP-kN)*48);
  k_zero<<<64, 256, 0, stream>>>((float*)(s_nbf + (size_t)kN*64), (kNP-kN)*32);
  k_zero<<<64, 256, 0, stream>>>(deg, kN);
  k_deg<<<EB, 256, 0, stream>>>(tgt, deg);
  k_edge_geom<<<EB, 256, 0, stream>>>(p_a, src, tgt, dbuf, abuf, rbuf);

  for (int i = 0; i < 5; i++){
    bool has_v = i > 0, mlp = i < 4;
    const LayerP& Lp = Ls[i];
    int out2 = 64 + (has_v ? 32 : 16) + 17;

    k_zero<<<64, 256, 0, stream>>>(gstat, kG*130);
    k_gstat<<<NB, 256, 0, stream>>>(s_a, v_a, batch, gstat);
    k_gnorm<<<NB, 256, 0, stream>>>(s_a, v_a, batch, gstat, Lp.ln_w, Lp.ln_b, s_n, s_nbf, v_n);

    PackArgs pa{}; int c = 0;
    auto add = [&](const float* sp, int K, int Nn, int Kt, int Nt, int off){
      pa.e[c].src = sp; pa.e[c].K = K; pa.e[c].Nn = Nn; pa.e[c].Kt = Kt; pa.e[c].Nt = Nt; pa.e[c].off = off; c++;
    };
    add(Lp.We_pre, 16, 16, 1, 1, O_WPRE);
    add(Lp.We1, 167, 64, 6, 4, O_W1);
    add(Lp.We2, 64, out2, 2, 8, O_W2);
    add(Lp.We_post, 16, 16, 1, 1, O_WPOST);
    add(Lp.Ws, 64, 64, 2, 4, O_WS);
    if (has_v) add(Lp.Wv, 16, 16, 1, 1, O_WV);
    if (mlp){
      add(Lp.Wu1, 16, 16, 1, 1, O_WU1); add(Lp.Wu2, 16, 16, 1, 1, O_WU2);
      add(Lp.Wg1, 80, 64, 3, 4, O_WG1); add(Lp.Wg2, 64, 80, 2, 5, O_WG2);
    }
    pa.count = c; pa.dst = wfrag;
    k_pack<<<64, 256, 0, stream>>>(pa);

    NodePreArgs np{ s_nbf, v_n, wfrag + O_WS, Lp.bs, wfrag + O_WV, has_v ? 1 : 0, ns, vv };
    k_nodepre<<<nodeBlk, 64, 0, stream>>>(np);

    k_zero<<<256, 256, 0, stream>>>(s_sum, kN*115);

    EdgeArgs ea{ src, tgt, dbuf, abuf, rbuf, ohe, gemb, s_nbf, e_a, ns, vv,
                 s_sum, v_sum, p_sum,
                 wfrag + O_WPRE, wfrag + O_W1, wfrag + O_W2, wfrag + O_WPOST,
                 Lp.be_pre, Lp.be1, Lp.be2, Lp.be_post,
                 has_v ? 1 : 0, out2 };
    k_edge<<<edgeBlk, 64, 0, stream>>>(ea);

    k_node_update<<<NB, 256, 0, stream>>>(s_n, v_n, s_sum, v_sum, p_sum, deg, s_a, v_a, p_a);

    if (mlp){
      MlpArgs ma{ s_a, v_a, wfrag + O_WU1, wfrag + O_WU2, wfrag + O_WG1, wfrag + O_WG2,
                  Lp.bg1, Lp.bg2 };
      k_node_mlp<<<nodeBlk, 64, 0, stream>>>(ma);
    }
    if (i < 4) k_edge_geom<<<EB, 256, 0, stream>>>(p_a, src, tgt, dbuf, abuf, rbuf);
  }

  // final graph norm -> outputs
  k_zero<<<64, 256, 0, stream>>>(gstat, kG*130);
  k_gstat<<<NB, 256, 0, stream>>>(s_a, v_a, batch, gstat);
  k_gnorm<<<NB, 256, 0, stream>>>(s_a, v_a, batch, gstat, out_ln_w, out_ln_b, out_s, s_nbf, out_v);
  hipMemcpyAsync(out_e, e_a, (size_t)kE*16*4, hipMemcpyDeviceToDevice, stream);
  hipMemcpyAsync(out_p, p_a, (size_t)kN*3*4,  hipMemcpyDeviceToDevice, stream);
}
